// MultiHeadAttention_32581621908200
// MI455X (gfx1250) — compile-verified, matrix operand order checked by reading, not run on hardware
//
#include <hip/hip_runtime.h>
#include <hip/hip_bf16.h>

// ---------------------------------------------------------------------------
// MultiHeadAttention for MI455X (gfx1250): bf16 WMMA GEMMs + flash attention,
// with Tensor Data Mover (TDM) staging of Q/K tiles when available.
// ---------------------------------------------------------------------------

typedef __bf16 bf16_t;
typedef __attribute__((ext_vector_type(16))) __bf16 v16bf;
typedef __attribute__((ext_vector_type(8)))  __bf16 v8bf;
typedef __attribute__((ext_vector_type(4)))  __bf16 v4bf;
typedef __attribute__((ext_vector_type(8)))  float  v8f;
typedef __attribute__((ext_vector_type(4)))  float  v4f;
typedef __attribute__((ext_vector_type(4)))  unsigned int v4u;
typedef __attribute__((ext_vector_type(8)))  int    v8i;
typedef __attribute__((ext_vector_type(4)))  int    v4i;

constexpr int kDim   = 1024;
constexpr int kHeads = 16;
constexpr int kDh    = 64;     // head dim
constexpr int kSeq   = 2048;
constexpr int kBatch = 2;
constexpr int kM     = kBatch * kSeq;  // 4096 rows
constexpr int kNKV   = 64;             // KV tile width in attention

#if defined(__HIP_DEVICE_COMPILE__) &&                         \
    __has_builtin(__builtin_amdgcn_tensor_load_to_lds) &&      \
    __has_builtin(__builtin_amdgcn_s_wait_tensorcnt)
#define USE_TDM 1
#else
#define USE_TDM 0
#endif

// ---------------------------------------------------------------------------
// WMMA helpers
// ---------------------------------------------------------------------------

static __device__ __forceinline__ v8f wmma_bf16(v16bf a, v16bf b, v8f c) {
  // D = A(16x32) * B(32x16) + C, f32 accumulate
  return __builtin_amdgcn_wmma_f32_16x16x32_bf16(
      /*neg_a=*/false, a, /*neg_b=*/false, b,
      /*c_mod=*/(short)0, c, /*reuse_a=*/false, /*reuse_b=*/false);
}

// Load a 16x32 (bf16) A/B fragment slice for one lane from an LDS row.
// Per ISA layout: element e holds K = (e<8 ? e : e+8) + 8*g, g = lane>>4.
// => two contiguous 8-half (16B) chunks at byte offsets (g*8)*2 and (16+g*8)*2.
static __device__ __forceinline__ v16bf load_frag(const bf16_t* row, int g) {
  v8bf lo = *reinterpret_cast<const v8bf*>(row + g * 8);
  v8bf hi = *reinterpret_cast<const v8bf*>(row + 16 + g * 8);
  return __builtin_shufflevector(lo, hi, 0, 1, 2, 3, 4, 5, 6, 7,
                                 8, 9, 10, 11, 12, 13, 14, 15);
}

// ---------------------------------------------------------------------------
// Tensor Data Mover: 2D bf16 tile load (global -> LDS) with hardware padding.
// Produces LDS layout [tile_h][tile_w + 8] halves (pad 16B per 128B row),
// which requires tile_w == 64 halves (row = 32 DWORDs).
// D# packing per CDNA5 ISA ch.8 (group0 128b, group1 256b; groups 2/3 zero).
// ---------------------------------------------------------------------------

#if USE_TDM
static __device__ __forceinline__ void tdm_load_tile_2d_bf16(
    const void* gaddr, unsigned lds_byte_addr, int tile_w_elems,
    int tile_h_rows, int row_stride_elems) {
  unsigned long long ga = (unsigned long long)(size_t)gaddr;
  v4u g0;
  g0[0] = 1u;                                   // count=1, user descriptor
  g0[1] = lds_byte_addr;                        // LDS dest (byte address)
  g0[2] = (unsigned)(ga & 0xffffffffu);         // global_addr[31:0]
  g0[3] = (unsigned)((ga >> 32) & 0x1ffffffu)   // global_addr[56:32]
        | (2u << 30);                           // type=2 ("image")
  v8i g1;
  // data_size=2B (code 1), pad_enable, pad_interval=32 DW (code 4),
  // pad_amount=4 DW (code 3), workgroup_mask=0 (not in a cluster).
  g1[0] = (1 << 16) | (1 << 20) | (4 << 22) | (3 << 25);
  const int tw = tile_w_elems, th = tile_h_rows;
  g1[1] = (tw & 0xffff) << 16;                          // tensor_dim0[15:0]
  g1[2] = ((tw >> 16) & 0xffff) | ((th & 0xffff) << 16);// dim0 hi | dim1 lo
  g1[3] = ((th >> 16) & 0xffff) | ((tw & 0xffff) << 16);// dim1 hi | tile_dim0
  g1[4] = (th & 0xffff);                                // tile_dim1 (tile_dim2=0)
  g1[5] = row_stride_elems;                             // tensor_dim0_stride lo
  g1[6] = 0;                                            // stride hi | dim1_stride lo
  g1[7] = 0;
  v4i z4 = {0, 0, 0, 0};
  v8i z8 = {0, 0, 0, 0, 0, 0, 0, 0};
  // 6-arg toolchain variant: (g0, g1, g2, g3, g4, cpol)
  __builtin_amdgcn_tensor_load_to_lds(g0, g1, z4, z4, z8, 0);
}
#endif

// ---------------------------------------------------------------------------
// Tile staging: global (f32 or bf16) -> LDS bf16, 16B vector traffic.
// Tiles are rows x 32 columns; LDS row stride in halves = sstride.
// ---------------------------------------------------------------------------

static __device__ __forceinline__ void copy_tile(const float* __restrict__ g,
                                                 int gstride, bf16_t* s,
                                                 int sstride, int rows,
                                                 int tid) {
  // 32 f32 cols = 8 chunks of 4 floats per row
  for (int c = tid; c < rows * 8; c += 256) {
    int r = c >> 3, cc = (c & 7) * 4;
    v4f v = *reinterpret_cast<const v4f*>(g + (size_t)r * gstride + cc);
    v4bf o = {(bf16_t)v[0], (bf16_t)v[1], (bf16_t)v[2], (bf16_t)v[3]};
    *reinterpret_cast<v4bf*>(s + (size_t)r * sstride + cc) = o;
  }
}

static __device__ __forceinline__ void copy_tile(const bf16_t* __restrict__ g,
                                                 int gstride, bf16_t* s,
                                                 int sstride, int rows,
                                                 int tid) {
  // 32 bf16 cols = 4 chunks of 8 halves per row
  for (int c = tid; c < rows * 4; c += 256) {
    int r = c >> 2, cc = (c & 3) * 8;
    *reinterpret_cast<v8bf*>(s + (size_t)r * sstride + cc) =
        *reinterpret_cast<const v8bf*>(g + (size_t)r * gstride + cc);
  }
}

// ---------------------------------------------------------------------------
// GEMM: C[M,N] = A[M,K] * W[N,K]^T + bias[N]
// A: f32 or bf16. Output: bf16 (QKV) or f32 (final projection).
// Block tile 128x128, 256 threads = 8 waves in a 2x4 grid; each wave 64x32.
// ---------------------------------------------------------------------------

template <typename AT, typename OT>
__global__ __launch_bounds__(256) void gemm_xwt_bias(
    const AT* __restrict__ A, const float* __restrict__ W,
    const float* __restrict__ bias, OT* __restrict__ C, int M, int N, int K) {
  __shared__ alignas(16) bf16_t sA[128][40];  // 32 + 8 pad halves
  __shared__ alignas(16) bf16_t sB[128][40];

  const int tid  = threadIdx.x;
  const int lane = tid & 31;
  const int wid  = tid >> 5;
  const int g    = lane >> 4;
  const int ln   = lane & 15;
  const int wm   = wid & 1;   // 2 wave-rows  (64 rows each)
  const int wn   = wid >> 1;  // 4 wave-cols  (32 cols each)
  const int m0   = blockIdx.x * 128;
  const int n0   = blockIdx.y * 128;

  v8f acc[4][2] = {};

  for (int k0 = 0; k0 < K; k0 += 32) {
    __syncthreads();
    copy_tile(A + (size_t)m0 * K + k0, K, &sA[0][0], 40, 128, tid);
    copy_tile(W + (size_t)n0 * K + k0, K, &sB[0][0], 40, 128, tid);
    if (k0 + 32 < K) {  // warm caches for next K tile (global_prefetch_b8)
      __builtin_prefetch(A + (size_t)(m0 + (tid & 127)) * K + k0 + 32, 0, 1);
      __builtin_prefetch(W + (size_t)(n0 + (tid & 127)) * K + k0 + 32, 0, 1);
    }
    __syncthreads();

    v16bf aF[4], bF[2];
#pragma unroll
    for (int mi = 0; mi < 4; ++mi)
      aF[mi] = load_frag(&sA[wm * 64 + mi * 16 + ln][0], g);
#pragma unroll
    for (int ni = 0; ni < 2; ++ni)
      bF[ni] = load_frag(&sB[wn * 32 + ni * 16 + ln][0], g);
#pragma unroll
    for (int mi = 0; mi < 4; ++mi)
#pragma unroll
      for (int ni = 0; ni < 2; ++ni)
        acc[mi][ni] = wmma_bf16(aF[mi], bF[ni], acc[mi][ni]);
  }

  // Epilogue: bias add + store.  C layout: element i @ (lane half g) = row i+8g.
#pragma unroll
  for (int ni = 0; ni < 2; ++ni) {
    const int col = n0 + wn * 32 + ni * 16 + ln;
    const float bvn = bias[col];
#pragma unroll
    for (int mi = 0; mi < 4; ++mi) {
#pragma unroll
      for (int i = 0; i < 8; ++i) {
        const int row = m0 + wm * 64 + mi * 16 + i + 8 * g;
        C[(size_t)row * N + col] = (OT)(acc[mi][ni][i] + bvn);
      }
    }
  }
}

// ---------------------------------------------------------------------------
// Flash attention: one block per (qtile=128, head, batch), 256 threads.
// Wave w owns 16 query rows; KV processed in 64-wide tiles with online
// softmax (fp32) between the QK^T and PV WMMAs.  Q/K tiles are staged by
// the Tensor Data Mover (TENSORcnt) while V is transposed by VALU threads.
// ---------------------------------------------------------------------------

static __device__ __forceinline__ float half_reduce_max(float v) {
#pragma unroll
  for (int m = 1; m < 16; m <<= 1) v = fmaxf(v, __shfl_xor(v, m, 32));
  return v;
}
static __device__ __forceinline__ float half_reduce_sum(float v) {
#pragma unroll
  for (int m = 1; m < 16; m <<= 1) v += __shfl_xor(v, m, 32);
  return v;
}

__global__ __launch_bounds__(256) void attn_kernel(
    const bf16_t* __restrict__ Qb, const bf16_t* __restrict__ Kb,
    const bf16_t* __restrict__ Vb, bf16_t* __restrict__ Ob) {
  __shared__ alignas(16) bf16_t sQ[128][72];        // 128 x 64 (+8 pad)
  __shared__ alignas(16) bf16_t sK[kNKV][72];       //  64 x 64 (+8 pad)
  __shared__ alignas(16) bf16_t sVt[kDh][kNKV + 8]; // V transposed: [d][kk]
  __shared__ alignas(16) bf16_t sP[8][16][kNKV + 8];// per-wave probs

  const int tid  = threadIdx.x;
  const int lane = tid & 31;
  const int wid  = tid >> 5;
  const int g    = lane >> 4;
  const int ln   = lane & 15;

  const int q0 = blockIdx.x * 128;
  const int h  = blockIdx.y;
  const int b  = blockIdx.z;

  const bf16_t* Qg = Qb + (size_t)(b * kSeq) * kDim + h * kDh;
  const bf16_t* Kg = Kb + (size_t)(b * kSeq) * kDim + h * kDh;
  const bf16_t* Vg = Vb + (size_t)(b * kSeq) * kDim + h * kDh;

  // Stage the 128x64 Q tile once (TDM when available; awaited with K below).
#if USE_TDM
  if (wid == 0) {
    tdm_load_tile_2d_bf16(Qg + (size_t)q0 * kDim,
                          (unsigned)(size_t)&sQ[0][0], kDh, 128, kDim);
  }
#else
  for (int c = tid; c < 128 * 8; c += 256) {
    int r = c >> 3, cc = (c & 7) * 8;
    *reinterpret_cast<v8bf*>(&sQ[r][cc]) =
        *reinterpret_cast<const v8bf*>(Qg + (size_t)(q0 + r) * kDim + cc);
  }
#endif

  float mrow[8], lrow[8];
  v8f   oacc[4] = {};
#pragma unroll
  for (int i = 0; i < 8; ++i) { mrow[i] = -1e30f; lrow[i] = 0.f; }

  const float scale = 0.125f;  // 1/sqrt(64)

  for (int k0 = 0; k0 < kSeq; k0 += kNKV) {
    __syncthreads();  // previous iteration done with sK/sVt/sP

#if USE_TDM
    if (wid == 0) {
      tdm_load_tile_2d_bf16(Kg + (size_t)k0 * kDim,
                            (unsigned)(size_t)&sK[0][0], kDh, kNKV, kDim);
    }
#endif
    // Stage V tile transposed (and K manually if no TDM) while DMA runs.
    for (int c = tid; c < kNKV * 8; c += 256) {
      int r = c >> 3, cc = (c & 7) * 8;
      v8bf v = *reinterpret_cast<const v8bf*>(Vg + (size_t)(k0 + r) * kDim + cc);
#pragma unroll
      for (int j = 0; j < 8; ++j) sVt[cc + j][r] = v[j];
#if !USE_TDM
      *reinterpret_cast<v8bf*>(&sK[r][cc]) =
          *reinterpret_cast<const v8bf*>(Kg + (size_t)(k0 + r) * kDim + cc);
#endif
    }
#if USE_TDM
    if (wid == 0) __builtin_amdgcn_s_wait_tensorcnt(0);
#endif
    __syncthreads();

    // energy = Q(16x64) * K^T(64x64): B fragment = K rows (K-major already).
    v8f eacc[4] = {};
#pragma unroll
    for (int ks = 0; ks < kDh; ks += 32) {
      v16bf aF = load_frag(&sQ[wid * 16 + ln][ks], g);
#pragma unroll
      for (int j = 0; j < 4; ++j) {
        v16bf bF = load_frag(&sK[j * 16 + ln][ks], g);
        eacc[j] = wmma_bf16(aF, bF, eacc[j]);
      }
    }

    // Online softmax (rows live in 16-lane halves; element i => row i+8g).
#pragma unroll
    for (int i = 0; i < 8; ++i) {
      float mx = -1e30f;
#pragma unroll
      for (int j = 0; j < 4; ++j) mx = fmaxf(mx, eacc[j][i]);
      mx = half_reduce_max(mx) * scale;
      const float mnew  = fmaxf(mrow[i], mx);
      const float alpha = __expf(mrow[i] - mnew);
      float rsum = 0.f;
#pragma unroll
      for (int j = 0; j < 4; ++j) {
        const float p = __expf(eacc[j][i] * scale - mnew);
        rsum += p;
        sP[wid][i + 8 * g][j * 16 + ln] = (bf16_t)p;
      }
      rsum   = half_reduce_sum(rsum);
      lrow[i] = lrow[i] * alpha + rsum;
      mrow[i] = mnew;
#pragma unroll
      for (int d = 0; d < 4; ++d) oacc[d][i] *= alpha;
    }
    __syncthreads();  // sP visible (and all eacc consumers done with sK)

    // out += P(16x64) * V(64x64): B fragment = sVt rows (d-major).
#pragma unroll
    for (int ks = 0; ks < kNKV; ks += 32) {
      v16bf aF = load_frag(&sP[wid][ln][ks], g);
#pragma unroll
      for (int d = 0; d < 4; ++d) {
        v16bf bF = load_frag(&sVt[d * 16 + ln][ks], g);
        oacc[d] = wmma_bf16(aF, bF, oacc[d]);
      }
    }
  }

  // Normalize and write [b, s, h*64+d] so the final GEMM reads row-major.
#pragma unroll
  for (int d = 0; d < 4; ++d) {
#pragma unroll
    for (int i = 0; i < 8; ++i) {
      const int r  = q0 + wid * 16 + i + 8 * g;
      const int cc = h * kDh + d * 16 + ln;
      const float v = oacc[d][i] / lrow[i];
      Ob[(size_t)(b * kSeq + r) * kDim + cc] = (bf16_t)v;
    }
  }
}

// ---------------------------------------------------------------------------
// Host launcher
// ---------------------------------------------------------------------------

extern "C" void kernel_launch(void* const* d_in, const int* in_sizes, int n_in,
                              void* d_out, int out_size, void* d_ws,
                              size_t ws_size, hipStream_t stream) {
  const float* query = (const float*)d_in[0];
  const float* key   = (const float*)d_in[1];
  const float* value = (const float*)d_in[2];
  const float* wq    = (const float*)d_in[3];
  const float* bq    = (const float*)d_in[4];
  const float* wk    = (const float*)d_in[5];
  const float* bk    = (const float*)d_in[6];
  const float* wv    = (const float*)d_in[7];
  const float* bv    = (const float*)d_in[8];
  const float* wo    = (const float*)d_in[9];
  const float* bo    = (const float*)d_in[10];

  // Workspace: Q, K, V, attn-out as bf16 [4096][1024]  (4 x 8 MB = 32 MB)
  bf16_t* Qb = (bf16_t*)d_ws;
  bf16_t* Kb = Qb + (size_t)kM * kDim;
  bf16_t* Vb = Kb + (size_t)kM * kDim;
  bf16_t* Ab = Vb + (size_t)kM * kDim;

  dim3 gp(kM / 128, kDim / 128, 1);
  gemm_xwt_bias<float, bf16_t><<<gp, 256, 0, stream>>>(query, wq, bq, Qb, kM, kDim, kDim);
  gemm_xwt_bias<float, bf16_t><<<gp, 256, 0, stream>>>(key,   wk, bk, Kb, kM, kDim, kDim);
  gemm_xwt_bias<float, bf16_t><<<gp, 256, 0, stream>>>(value, wv, bv, Vb, kM, kDim, kDim);

  dim3 ga(kSeq / 128, kHeads, kBatch);
  attn_kernel<<<ga, 256, 0, stream>>>(Qb, Kb, Vb, Ab);

  gemm_xwt_bias<bf16_t, float><<<gp, 256, 0, stream>>>(Ab, wo, bo, (float*)d_out,
                                                       kM, kDim, kDim);
}
